// SecondGCN_9749575762779
// MI455X (gfx1250) — matrix operand order repaired and checked by reading, not compiled, not run on hardware
//
#include <hip/hip_runtime.h>
#include <cstdint>
#include <cstddef>

typedef __bf16 bf16;
typedef bf16  v16bf __attribute__((ext_vector_type(16)));
typedef bf16  v8bf  __attribute__((ext_vector_type(8)));
typedef float v8f   __attribute__((ext_vector_type(8)));

#define E_REL 250000
#define FEAT  768
#define BM    128
#define BN    128
#define BK    32
#define A_STRIDE 40   // bf16 elems per A row in LDS (80 B, 16B-aligned, 20-bank stride)
#define B_STRIDE 40   // bf16 elems per B column in LDS (fragment-order layout)

// ---------------------------------------------------------------------------
// Utility kernels
// ---------------------------------------------------------------------------
__global__ __launch_bounds__(256) void zero4_kernel(float4* p, long n4) {
    long i = (long)blockIdx.x * 256 + threadIdx.x;
    if (i < n4) p[i] = make_float4(0.f, 0.f, 0.f, 0.f);
}

__global__ __launch_bounds__(256) void count_deg_kernel(
    const int* __restrict__ src, const int* __restrict__ dst,
    float* __restrict__ outd, float* __restrict__ ind, int E) {
    int e = blockIdx.x * 256 + threadIdx.x;
    if (e < E) {
        atomicAdd(&outd[src[e]], 1.0f);
        atomicAdd(&ind[dst[e]], 1.0f);
    }
}

__global__ __launch_bounds__(256) void deg_to_scale_kernel(float* __restrict__ p, int n) {
    int i = blockIdx.x * 256 + threadIdx.x;
    if (i < n) p[i] = rsqrtf(fmaxf(p[i], 1.0f));   // clamp(deg,1)^-0.5
}

// Per-destination-type bias sums. RELS dst types:
//   M: M_M(0), E_M(4), S_M(5);  E: M_E(1), S_E(6), E_E(8);  S: M_S(2), E_S(3), S_S(7)
__global__ __launch_bounds__(256) void bias_sums_kernel(
    const float* b0, const float* b1, const float* b2, const float* b3,
    const float* b4, const float* b5, const float* b6, const float* b7,
    const float* b8, float* __restrict__ B) {
    int f = blockIdx.x * 256 + threadIdx.x;
    if (f < FEAT) {
        B[f]            = b0[f] + b4[f] + b5[f];
        B[FEAT + f]     = b1[f] + b6[f] + b8[f];
        B[2 * FEAT + f] = b2[f] + b3[f] + b7[f];
    }
}

// ---------------------------------------------------------------------------
// bf16 WMMA GEMM:  H[nRows,768] = (X * rowScale) @ W   (K = N = 768)
// 256 threads = 8 waves; block tile 128x128; wave tile 32x64 (2x4 fragments).
// A LDS: row-major, stride 40 bf16  -> A fragment = 2x ds_load_b128
// B LDS: fragment-order per column (32 K-values: [k0..15][k16..31]),
//        stride 40 bf16             -> B fragment = 2x ds_load_b128
// OOB rows: clamp the address (always-valid load) and zero the row scale.
// C store: block-uniform fast path (no guards) for interior blocks.
// ---------------------------------------------------------------------------
__global__ __launch_bounds__(256) void gemm_xw_bf16_kernel(
    const float* __restrict__ X, const float* __restrict__ Wm,
    const float* __restrict__ rowScale, float* __restrict__ H, int nRows) {
    __shared__ bf16 As[BM * A_STRIDE];   // 10240 B
    __shared__ bf16 Bs[BN * B_STRIDE];   // 10240 B

    const int tid  = threadIdx.x;
    const int lane = tid & 31;
    const int wave = tid >> 5;
    const int wm   = wave >> 1;      // 0..3 -> 32-row group
    const int wn   = wave & 1;       // 0..1 -> 64-col group
    const int bm   = blockIdx.x * BM;
    const int bn   = blockIdx.y * BN;

    // A staging: thread -> (row = tid/2, 16 cols at (tid&1)*16)
    const int  arow   = tid >> 1;
    const int  acol   = (tid & 1) * 16;
    const int  rowG   = bm + arow;
    const int  rowC   = (rowG < nRows) ? rowG : (nRows - 1);   // clamped, always valid
    const float ascale = (rowG < nRows) ? rowScale[rowC] : 0.f; // zero kills OOB rows
    const float4* aptr = (const float4*)(X + (size_t)rowC * FEAT);

    // B staging: thread -> column nloc = tid&127, K-half hB = tid>>7
    const int nloc = tid & 127;
    const int hB   = tid >> 7;

    const int m     = lane & 15;     // fragment row / col within 16
    const int halfk = lane >> 4;     // half-wave select

    v8f acc[2][4] = {};

    for (int k0 = 0; k0 < FEAT; k0 += BK) {
        // ---- stage A tile (fp32 -> bf16, fold row scale), row-major ----
        {
            v8bf p0, p1;
            #pragma unroll
            for (int q = 0; q < 2; ++q) {
                float4 v0 = aptr[(k0 + acol) / 4 + 2 * q];
                float4 v1 = aptr[(k0 + acol) / 4 + 2 * q + 1];
                v8bf* pk = q ? &p1 : &p0;
                (*pk)[0] = (bf16)(v0.x * ascale); (*pk)[1] = (bf16)(v0.y * ascale);
                (*pk)[2] = (bf16)(v0.z * ascale); (*pk)[3] = (bf16)(v0.w * ascale);
                (*pk)[4] = (bf16)(v1.x * ascale); (*pk)[5] = (bf16)(v1.y * ascale);
                (*pk)[6] = (bf16)(v1.z * ascale); (*pk)[7] = (bf16)(v1.w * ascale);
            }
            *(v8bf*)&As[arow * A_STRIDE + acol]     = p0;
            *(v8bf*)&As[arow * A_STRIDE + acol + 8] = p1;
        }
        // ---- stage B tile (fp32 -> bf16) in fragment order ----
        // element e of column nloc, half hB:  W[(k0 + hB*16 + e)*768 + bn + nloc]
        {
            const float* wp = Wm + (size_t)(k0 + hB * 16) * FEAT + bn + nloc;
            v8bf p0, p1;
            #pragma unroll
            for (int e = 0; e < 8; ++e) p0[e] = (bf16)wp[(size_t)e * FEAT];
            #pragma unroll
            for (int e = 0; e < 8; ++e) p1[e] = (bf16)wp[(size_t)(e + 8) * FEAT];
            *(v8bf*)&Bs[nloc * B_STRIDE + hB * 16]     = p0;
            *(v8bf*)&Bs[nloc * B_STRIDE + hB * 16 + 8] = p1;
        }
        __syncthreads();

        // ---- A fragments: 2x b128 each (K = 16h+e) ----
        v16bf af[2];
        #pragma unroll
        for (int mi = 0; mi < 2; ++mi) {
            const bf16* ar = &As[(wm * 32 + mi * 16 + m) * A_STRIDE + 16 * halfk];
            v8bf lo = *(const v8bf*)(ar);
            v8bf hi = *(const v8bf*)(ar + 8);
            af[mi] = __builtin_shufflevector(lo, hi, 0, 1, 2, 3, 4, 5, 6, 7,
                                             8, 9, 10, 11, 12, 13, 14, 15);
        }
        // ---- B fragments (K = 16*half + e) + WMMA ----
        #pragma unroll
        for (int ni = 0; ni < 4; ++ni) {
            const bf16* br = &Bs[(wn * 64 + ni * 16 + m) * B_STRIDE + 16 * halfk];
            v8bf lo = *(const v8bf*)(br);
            v8bf hi = *(const v8bf*)(br + 8);
            v16bf bfr = __builtin_shufflevector(lo, hi, 0, 1, 2, 3, 4, 5, 6, 7,
                                                8, 9, 10, 11, 12, 13, 14, 15);
            acc[0][ni] = __builtin_amdgcn_wmma_f32_16x16x32_bf16(
                false, af[0], false, bfr, (short)0, acc[0][ni], false, false);
            acc[1][ni] = __builtin_amdgcn_wmma_f32_16x16x32_bf16(
                false, af[1], false, bfr, (short)0, acc[1][ni], false, false);
        }
        __syncthreads();
    }

    // ---- store C fragments (ISA 32-bit C/D layout) ----
    if (bm + BM <= nRows) {
        // interior block: straight-line stores, no guards
        #pragma unroll
        for (int mi = 0; mi < 2; ++mi) {
            int rbase = bm + wm * 32 + mi * 16 + halfk * 8;
            #pragma unroll
            for (int ni = 0; ni < 4; ++ni) {
                int col = bn + wn * 64 + ni * 16 + m;
                float* hp = H + (size_t)rbase * FEAT + col;
                #pragma unroll
                for (int i = 0; i < 8; ++i)
                    hp[(size_t)i * FEAT] = acc[mi][ni][i];
            }
        }
    } else {
        // boundary block: per-row guard
        #pragma unroll
        for (int mi = 0; mi < 2; ++mi) {
            int rbase = bm + wm * 32 + mi * 16 + halfk * 8;
            #pragma unroll
            for (int ni = 0; ni < 4; ++ni) {
                int col = bn + wn * 64 + ni * 16 + m;
                #pragma unroll
                for (int i = 0; i < 8; ++i) {
                    int row = rbase + i;
                    if (row < nRows) H[(size_t)row * FEAT + col] = acc[mi][ni][i];
                }
            }
        }
    }
}

// ---------------------------------------------------------------------------
// Edge scatter: one wave per edge; acc[dst] += h[src] * (w[e] * inScale[dst])
// ---------------------------------------------------------------------------
__global__ __launch_bounds__(256) void edge_scatter_kernel(
    const float* __restrict__ H, const int* __restrict__ src,
    const int* __restrict__ dst, const float* __restrict__ w,
    const float* __restrict__ inScale, float* __restrict__ acc, int E) {
    int warp = threadIdx.x >> 5;
    int lane = threadIdx.x & 31;
    int e = blockIdx.x * 8 + warp;
    if (e >= E) return;
    int s = src[e], d = dst[e];
    float c = w[e] * inScale[d];
    const float4* hr = (const float4*)(H + (size_t)s * FEAT);
    float* ar = acc + (size_t)d * FEAT;
    #pragma unroll
    for (int i = 0; i < 6; ++i) {            // 6 * 32 lanes * 4 = 768 feats
        int f4 = lane + i * 32;
        float4 v = hr[f4];
        int base = f4 * 4;
        atomicAdd(ar + base + 0, v.x * c);
        atomicAdd(ar + base + 1, v.y * c);
        atomicAdd(ar + base + 2, v.z * c);
        atomicAdd(ar + base + 3, v.w * c);
    }
}

// ---------------------------------------------------------------------------
// Final: out = relu(out + biasSum[type])
// ---------------------------------------------------------------------------
__global__ __launch_bounds__(256) void bias_relu_kernel(
    float* __restrict__ out, const float* __restrict__ B) {
    long i = (long)blockIdx.x * 256 + threadIdx.x;   // float4 index
    const long n4 = (long)160000 * FEAT / 4;         // 30,720,000
    if (i >= n4) return;
    long row = i / (FEAT / 4);
    int  f4  = (int)(i % (FEAT / 4));
    int  t   = (row < 100000) ? 0 : ((row < 140000) ? 1 : 2);
    float4 v  = ((float4*)out)[i];
    float4 bb = ((const float4*)B)[t * (FEAT / 4) + f4];
    v.x = fmaxf(v.x + bb.x, 0.f);
    v.y = fmaxf(v.y + bb.y, 0.f);
    v.z = fmaxf(v.z + bb.z, 0.f);
    v.w = fmaxf(v.w + bb.w, 0.f);
    ((float4*)out)[i] = v;
}

// ---------------------------------------------------------------------------
// Launcher
// ---------------------------------------------------------------------------
extern "C" void kernel_launch(void* const* d_in, const int* in_sizes, int n_in,
                              void* d_out, int out_size, void* d_ws, size_t ws_size,
                              hipStream_t stream) {
    (void)in_sizes; (void)n_in; (void)out_size; (void)ws_size;

    const float* x[3] = {(const float*)d_in[0], (const float*)d_in[1], (const float*)d_in[2]};
    const int nNodes[3] = {100000, 40000, 20000};
    // RELS: M_M, M_E, M_S, E_S, E_M, S_M, S_E, S_S, E_E  (M=0, E=1, S=2)
    static const int relS[9] = {0, 0, 0, 1, 1, 2, 2, 2, 1};
    static const int relD[9] = {0, 1, 2, 2, 0, 0, 1, 2, 1};
    const size_t outOff[3] = {0, (size_t)100000 * FEAT, (size_t)140000 * FEAT};

    float* out = (float*)d_out;
    char*  ws  = (char*)d_ws;
    float* h        = (float*)ws;                                   // 100000*768 f32 (307.2 MB)
    float* outScale = (float*)(ws + (size_t)100000 * FEAT * 4);     // 100000 f32
    float* inScale  = outScale + 100000;                            // 100000 f32
    float* biasSum  = inScale + 100000;                             // 3*768 f32

    // zero accumulator (= output buffer)
    {
        long n4 = (long)160000 * FEAT / 4;
        zero4_kernel<<<(unsigned)((n4 + 255) / 256), 256, 0, stream>>>((float4*)out, n4);
    }
    // per-dst-type bias sums
    bias_sums_kernel<<<3, 256, 0, stream>>>(
        (const float*)d_in[3 + 5 * 0 + 4], (const float*)d_in[3 + 5 * 1 + 4],
        (const float*)d_in[3 + 5 * 2 + 4], (const float*)d_in[3 + 5 * 3 + 4],
        (const float*)d_in[3 + 5 * 4 + 4], (const float*)d_in[3 + 5 * 5 + 4],
        (const float*)d_in[3 + 5 * 6 + 4], (const float*)d_in[3 + 5 * 7 + 4],
        (const float*)d_in[3 + 5 * 8 + 4], biasSum);

    for (int r = 0; r < 9; ++r) {
        const int*   src = (const int*)  d_in[3 + 5 * r + 0];
        const int*   dst = (const int*)  d_in[3 + 5 * r + 1];
        const float* wE  = (const float*)d_in[3 + 5 * r + 2];
        const float* Wm  = (const float*)d_in[3 + 5 * r + 3];
        const int ns = nNodes[relS[r]];
        const int nd = nNodes[relD[r]];

        // degrees -> scales
        zero4_kernel<<<(ns / 4 + 255) / 256, 256, 0, stream>>>((float4*)outScale, ns / 4);
        zero4_kernel<<<(nd / 4 + 255) / 256, 256, 0, stream>>>((float4*)inScale, nd / 4);
        count_deg_kernel<<<(E_REL + 255) / 256, 256, 0, stream>>>(src, dst, outScale, inScale, E_REL);
        deg_to_scale_kernel<<<(ns + 255) / 256, 256, 0, stream>>>(outScale, ns);
        deg_to_scale_kernel<<<(nd + 255) / 256, 256, 0, stream>>>(inScale, nd);

        // h = (x * outScale) @ W   (bf16 WMMA, f32 accumulate)
        dim3 g((ns + BM - 1) / BM, FEAT / BN);
        gemm_xw_bf16_kernel<<<g, 256, 0, stream>>>(x[relS[r]], Wm, outScale, h, ns);

        // acc[dst] += h[src] * w[e] * inScale[dst]
        edge_scatter_kernel<<<(E_REL + 7) / 8, 256, 0, stream>>>(
            h, src, dst, wE, inScale, out + outOff[relD[r]], E_REL);
    }

    // out = relu(out + biasSum[type])
    {
        long n4 = (long)160000 * FEAT / 4;
        bias_relu_kernel<<<(unsigned)((n4 + 255) / 256), 256, 0, stream>>>(out, biasSum);
    }
}
